// MultiHeadAttentionParallel_19490561590292
// MI455X (gfx1250) — matrix operand order repaired and checked by reading, not compiled, not run on hardware
//
#include <hip/hip_runtime.h>
#include <hip/hip_bf16.h>
#include <stdint.h>

// ---------------------------------------------------------------------------
// MultiHeadAttention (B=2, S=4096, D=512, H=8, HD=64) for gfx1250 (MI455X).
// All matmuls on v_wmma_f32_16x16x32_bf16 (wave32 WMMA); V tiles staged with
// GLOBAL_LOAD_ASYNC_TO_LDS_B128 when the toolchain exposes the builtin.
// ws layout (32 MiB): [Q bf16][K bf16][V bf16] head-split [B,H,S,HD], then
// ctx bf16 [B,S,D].
// ---------------------------------------------------------------------------

typedef __bf16 bf16_t;
typedef __attribute__((ext_vector_type(16))) __bf16 v16bf;
typedef __attribute__((ext_vector_type(8)))  float  v8f;
typedef __attribute__((ext_vector_type(4)))  int    v4i;
typedef __attribute__((address_space(1))) v4i as1_v4i;  // global int4
typedef __attribute__((address_space(3))) v4i as3_v4i;  // LDS int4

#define B_   2
#define S_   4096
#define D_   512
#define H_   8
#define HD_  64
#define MTOT (B_ * S_)                         // 8192
#define PLANE ((size_t)B_ * H_ * S_ * HD_)     // 4194304 elems

#ifndef __has_builtin
#define __has_builtin(x) 0
#endif
#if __has_builtin(__builtin_amdgcn_global_load_async_to_lds_b128)
#define HAVE_ASYNC_LDS 1
#else
#define HAVE_ASYNC_LDS 0
#endif
#if __has_builtin(__builtin_amdgcn_s_wait_asynccnt)
#define WAIT_ASYNC0() __builtin_amdgcn_s_wait_asynccnt(0)
#else
#define WAIT_ASYNC0() asm volatile("s_wait_asynccnt 0" ::: "memory")
#endif

__device__ __forceinline__ v8f v8f_zero() {
  v8f z = {0.f, 0.f, 0.f, 0.f, 0.f, 0.f, 0.f, 0.f};
  return z;
}

__device__ __forceinline__ v8f wmma_bf16(v16bf a, v16bf b, v8f c) {
  // (neg_a, A, neg_b, B, c_mod, C, reuse_a, reuse_b)
  return __builtin_amdgcn_wmma_f32_16x16x32_bf16(false, a, false, b,
                                                 (short)0, c, false, false);
}

// Load a 16-element bf16 fragment as two 16B chunks (b128 loads).
__device__ __forceinline__ v16bf ld16(const bf16_t* p0, const bf16_t* p1) {
  union { v16bf v; uint4 q[2]; } u;
  u.q[0] = *reinterpret_cast<const uint4*>(p0);
  u.q[1] = *reinterpret_cast<const uint4*>(p1);
  return u.v;
}

// ---------------------------------------------------------------------------
// Kernel 1: Q/K/V projections. blockIdx.z selects which projection.
// C[m,n] = X[m,:] @ W[:,n], fp32 in, bf16 out, head-split output layout.
// WG = 256 thr = 8 waves -> 64x128 tile; each wave owns a 32x32 subtile.
// ---------------------------------------------------------------------------
__global__ __launch_bounds__(256) void proj_qkv_kernel(
    const float* __restrict__ Xq, const float* __restrict__ Xk,
    const float* __restrict__ Xv, const float* __restrict__ Wq,
    const float* __restrict__ Wk, const float* __restrict__ Wv,
    bf16_t* __restrict__ qkv) {
  const int which = blockIdx.z;
  const float* X = (which == 0) ? Xq : (which == 1) ? Xk : Xv;
  const float* W = (which == 0) ? Wq : (which == 1) ? Wk : Wv;
  bf16_t* Y = qkv + (size_t)which * PLANE;

  __shared__ alignas(16) bf16_t Xs[64][40];    // [m][k], padded stride
  __shared__ alignas(16) bf16_t Wst[128][40];  // [n][k] (transposed)

  const int tid = threadIdx.x, lane = tid & 31, wave = tid >> 5;
  const int m0 = blockIdx.x * 64, nb = blockIdx.y * 128;
  const int wm = (wave >> 2) * 32, wn = (wave & 3) * 32;

  v8f c00 = v8f_zero(), c01 = v8f_zero(), c10 = v8f_zero(), c11 = v8f_zero();

  const int xr = tid >> 2, xk = (tid & 3) * 8;      // X staging: 8 elems/thr
  const int wkr = tid >> 3, wn0 = (tid & 7) * 16;   // W staging: 16 elems/thr

  for (int kb = 0; kb < D_; kb += 32) {
    __syncthreads();
    {
      const float* src = X + (size_t)(m0 + xr) * D_ + kb + xk;
#pragma unroll
      for (int i = 0; i < 8; ++i) Xs[xr][xk + i] = (bf16_t)src[i];
      const float* wsrc = W + (size_t)(kb + wkr) * D_ + nb + wn0;
#pragma unroll
      for (int i = 0; i < 16; ++i) Wst[wn0 + i][wkr] = (bf16_t)wsrc[i];
    }
    if (kb + 32 < D_) {  // stream ahead (global_prefetch_b8)
      __builtin_prefetch(X + (size_t)(m0 + xr) * D_ + kb + 32 + xk, 0, 0);
      __builtin_prefetch(W + (size_t)(kb + 32 + wkr) * D_ + nb + wn0, 0, 0);
    }
    __syncthreads();

    const int kba = (lane >> 4) * 16;  // A-frag K base per lane half
    const bf16_t* ar0 = &Xs[wm + (lane & 15)][0];
    const bf16_t* ar1 = &Xs[wm + 16 + (lane & 15)][0];
    v16bf a0 = ld16(ar0 + kba, ar0 + kba + 8);
    v16bf a1 = ld16(ar1 + kba, ar1 + kba + 8);

    const int kbb = (lane >> 4) * 16;  // B-frag K base per lane half
    const bf16_t* br0 = &Wst[wn + (lane & 15)][kbb];
    const bf16_t* br1 = &Wst[wn + 16 + (lane & 15)][kbb];
    v16bf b0 = ld16(br0, br0 + 8);
    v16bf b1 = ld16(br1, br1 + 8);

    c00 = wmma_bf16(a0, b0, c00);
    c01 = wmma_bf16(a0, b1, c01);
    c10 = wmma_bf16(a1, b0, c10);
    c11 = wmma_bf16(a1, b1, c11);
  }

  // Epilogue: write bf16 head-split [B,H,S,HD].
  const int nlo = lane & 15, rhalf = (lane >> 4) * 8;
#pragma unroll
  for (int fm = 0; fm < 2; ++fm) {
#pragma unroll
    for (int fn = 0; fn < 2; ++fn) {
      v8f c = (fm == 0) ? ((fn == 0) ? c00 : c01) : ((fn == 0) ? c10 : c11);
      const int n = nb + wn + fn * 16 + nlo;
      const int h = n >> 6, hd = n & 63;
#pragma unroll
      for (int r = 0; r < 8; ++r) {
        const int m = m0 + wm + fm * 16 + rhalf + r;
        const int bb = m >> 12, s = m & (S_ - 1);
        Y[(((size_t)bb * H_ + h) * S_ + s) * HD_ + hd] = (bf16_t)c[r];
      }
    }
  }
}

// ---------------------------------------------------------------------------
// Kernel 2: flash attention. One wave per 16-row q-tile; 8 waves per WG.
// grid = (S/128, B*H). Online softmax with running (m, l) per row.
// V tiles go global -> LDS via async-load (ASYNCcnt) when available.
// ---------------------------------------------------------------------------
__global__ __launch_bounds__(256) void flash_attn_kernel(
    const bf16_t* __restrict__ qkv, const int* __restrict__ valid_lens,
    bf16_t* __restrict__ ctx) {
  const int bh = blockIdx.y;
  const int b = bh >> 3, h = bh & 7;
  const int lane = threadIdx.x & 31, wave = threadIdx.x >> 5;
  const int q0 = blockIdx.x * 128 + wave * 16;
  const int valid = valid_lens[b];

  const bf16_t* Q = qkv + (size_t)bh * S_ * HD_;
  const bf16_t* K = qkv + PLANE + (size_t)bh * S_ * HD_;
  const bf16_t* V = qkv + 2 * PLANE + (size_t)bh * S_ * HD_;

  __shared__ alignas(16) bf16_t Pst[8][16][32];  // per-wave P transpose tile
  __shared__ alignas(16) bf16_t Vs[8][32][64];   // per-wave V tile [j][d]

  const int nlo = lane & 15;
  const int kba = (lane >> 4) * 16;  // A-frag K base
  const int kbb = (lane >> 4) * 16;  // B-frag K base

  // Q fragments for this wave's 16 rows: d 0..31 and d 32..63.
  const bf16_t* qr = Q + (size_t)(q0 + nlo) * HD_;
  const v16bf aq0 = ld16(qr + kba, qr + kba + 8);
  const v16bf aq1 = ld16(qr + 32 + kba, qr + 32 + kba + 8);

  v8f acc0 = v8f_zero(), acc1 = v8f_zero(), acc2 = v8f_zero(),
      acc3 = v8f_zero();
  float m_run[8], l_run[8];
#pragma unroll
  for (int r = 0; r < 8; ++r) { m_run[r] = -3.0e38f; l_run[r] = 0.f; }

  for (int j0 = 0; j0 < S_; j0 += 32) {
    // ---- kick off V tile staging (32 rows x 64 d, row-major in LDS) ----
    {
      const bf16_t* vr = V + (size_t)(j0 + lane) * HD_;  // lane = row j
#if HAVE_ASYNC_LDS
#pragma unroll
      for (int i = 0; i < 8; ++i) {
        __builtin_amdgcn_global_load_async_to_lds_b128(
            (as1_v4i*)(vr + i * 8),
            (as3_v4i*)&Vs[wave][lane][i * 8],
            0, 0);
      }
#else
      union { uint4 q[8]; bf16_t e[64]; } vld;
#pragma unroll
      for (int i = 0; i < 8; ++i)
        vld.q[i] = *reinterpret_cast<const uint4*>(vr + i * 8);
#pragma unroll
      for (int i = 0; i < 8; ++i)
        *reinterpret_cast<uint4*>(&Vs[wave][lane][i * 8]) = vld.q[i];
#endif
    }

    // ---- scores for two adjacent 16-col tiles: S = Q K^T * (1/8) ----
    v8f sc0 = v8f_zero(), sc1 = v8f_zero();
#pragma unroll
    for (int t = 0; t < 2; ++t) {
      const bf16_t* kr = K + (size_t)(j0 + t * 16 + nlo) * HD_;
      v16bf bk0 = ld16(kr + kbb, kr + kbb + 8);            // d 0..31
      v16bf bk1 = ld16(kr + 32 + kbb, kr + 32 + kbb + 8);  // d 32..63
      v8f c = v8f_zero();
      c = wmma_bf16(aq0, bk0, c);
      c = wmma_bf16(aq1, bk1, c);
      const bool ok = (j0 + t * 16 + nlo) < valid;  // masked_softmax fill
      v8f& sc = (t == 0) ? sc0 : sc1;
#pragma unroll
      for (int r = 0; r < 8; ++r) sc[r] = ok ? c[r] * 0.125f : -1.0e6f;
    }

    if (j0 + 32 < S_) {  // prefetch next K/V tiles (global_prefetch_b8)
      __builtin_prefetch(K + (size_t)(j0 + 32 + nlo) * HD_, 0, 0);
      __builtin_prefetch(V + (size_t)(j0 + 32 + lane) * HD_, 0, 0);
    }

    // ---- online softmax (row stats via 16-lane butterfly) ----
    float scale_o[8];
#pragma unroll
    for (int r = 0; r < 8; ++r) {
      float v = fmaxf(sc0[r], sc1[r]);
      v = fmaxf(v, __shfl_xor(v, 1));
      v = fmaxf(v, __shfl_xor(v, 2));
      v = fmaxf(v, __shfl_xor(v, 4));
      v = fmaxf(v, __shfl_xor(v, 8));
      const float mn = fmaxf(m_run[r], v);
      const bf16_t p0b = (bf16_t)__expf(sc0[r] - mn);
      const bf16_t p1b = (bf16_t)__expf(sc1[r] - mn);
      Pst[wave][(lane >> 4) * 8 + r][nlo] = p0b;
      Pst[wave][(lane >> 4) * 8 + r][16 + nlo] = p1b;
      float s = (float)p0b + (float)p1b;  // match bf16 P fed to the WMMA
      s += __shfl_xor(s, 1);
      s += __shfl_xor(s, 2);
      s += __shfl_xor(s, 4);
      s += __shfl_xor(s, 8);
      const float so = __expf(m_run[r] - mn);
      l_run[r] = l_run[r] * so + s;
      m_run[r] = mn;
      scale_o[r] = so;
    }
#pragma unroll
    for (int r = 0; r < 8; ++r) {
      acc0[r] *= scale_o[r];
      acc1[r] *= scale_o[r];
      acc2[r] *= scale_o[r];
      acc3[r] *= scale_o[r];
    }

    // Same-wave LDS RAW (P transpose) + async V tile before fragment reads.
    asm volatile("s_wait_dscnt 0" ::: "memory");
    WAIT_ASYNC0();

    // ---- O += P @ V ----
    const bf16_t* pr = &Pst[wave][nlo][0];
    v16bf ap = ld16(pr + kba, pr + kba + 8);
#pragma unroll
    for (int f = 0; f < 4; ++f) {
      // B-frag (32x16): element e -> j = kbb + e, column d = f*16 + nlo.
      union { v16bf v; bf16_t e[16]; } ub;
#pragma unroll
      for (int e2 = 0; e2 < 16; ++e2)
        ub.e[e2] = Vs[wave][kbb + e2][f * 16 + nlo];
      v8f& acc = (f == 0) ? acc0 : (f == 1) ? acc1 : (f == 2) ? acc2 : acc3;
      acc = wmma_bf16(ap, ub.v, acc);
    }
  }

  // ---- epilogue: normalize and write bf16 ctx [B,S,D], D-col = h*64+d ----
#pragma unroll
  for (int r = 0; r < 8; ++r) {
    const float inv = __builtin_amdgcn_rcpf(l_run[r]);
    const int srow = q0 + (lane >> 4) * 8 + r;
    bf16_t* orow = ctx + ((size_t)b * S_ + srow) * D_ + h * HD_;
    orow[0 * 16 + nlo] = (bf16_t)(acc0[r] * inv);
    orow[1 * 16 + nlo] = (bf16_t)(acc1[r] * inv);
    orow[2 * 16 + nlo] = (bf16_t)(acc2[r] * inv);
    orow[3 * 16 + nlo] = (bf16_t)(acc3[r] * inv);
  }
}

// ---------------------------------------------------------------------------
// Kernel 3: out = ctx @ Wo, bf16 A, fp32 W (cast), fp32 out [B,S,D].
// ---------------------------------------------------------------------------
__global__ __launch_bounds__(256) void out_proj_kernel(
    const bf16_t* __restrict__ Xc, const float* __restrict__ Wo,
    float* __restrict__ out) {
  __shared__ alignas(16) bf16_t Xs[64][40];
  __shared__ alignas(16) bf16_t Wst[128][40];

  const int tid = threadIdx.x, lane = tid & 31, wave = tid >> 5;
  const int m0 = blockIdx.x * 64, nb = blockIdx.y * 128;
  const int wm = (wave >> 2) * 32, wn = (wave & 3) * 32;

  v8f c00 = v8f_zero(), c01 = v8f_zero(), c10 = v8f_zero(), c11 = v8f_zero();

  const int xr = tid >> 2, xk = (tid & 3) * 8;
  const int wkr = tid >> 3, wn0 = (tid & 7) * 16;

  for (int kb = 0; kb < D_; kb += 32) {
    __syncthreads();
    {
      const uint4 xv = *reinterpret_cast<const uint4*>(
          Xc + (size_t)(m0 + xr) * D_ + kb + xk);
      *reinterpret_cast<uint4*>(&Xs[xr][xk]) = xv;
      const float* wsrc = Wo + (size_t)(kb + wkr) * D_ + nb + wn0;
#pragma unroll
      for (int i = 0; i < 16; ++i) Wst[wn0 + i][wkr] = (bf16_t)wsrc[i];
    }
    if (kb + 32 < D_) {
      __builtin_prefetch(Xc + (size_t)(m0 + xr) * D_ + kb + 32 + xk, 0, 0);
      __builtin_prefetch(Wo + (size_t)(kb + 32 + wkr) * D_ + nb + wn0, 0, 0);
    }
    __syncthreads();

    const int kba = (lane >> 4) * 16;
    const bf16_t* ar0 = &Xs[wm + (lane & 15)][0];
    const bf16_t* ar1 = &Xs[wm + 16 + (lane & 15)][0];
    v16bf a0 = ld16(ar0 + kba, ar0 + kba + 8);
    v16bf a1 = ld16(ar1 + kba, ar1 + kba + 8);

    const int kbb = (lane >> 4) * 16;
    const bf16_t* br0 = &Wst[wn + (lane & 15)][kbb];
    const bf16_t* br1 = &Wst[wn + 16 + (lane & 15)][kbb];
    v16bf b0 = ld16(br0, br0 + 8);
    v16bf b1 = ld16(br1, br1 + 8);

    c00 = wmma_bf16(a0, b0, c00);
    c01 = wmma_bf16(a0, b1, c01);
    c10 = wmma_bf16(a1, b0, c10);
    c11 = wmma_bf16(a1, b1, c11);
  }

  const int nlo = lane & 15, rhalf = (lane >> 4) * 8;
#pragma unroll
  for (int fm = 0; fm < 2; ++fm) {
#pragma unroll
    for (int fn = 0; fn < 2; ++fn) {
      v8f c = (fm == 0) ? ((fn == 0) ? c00 : c01) : ((fn == 0) ? c10 : c11);
      const int n = nb + wn + fn * 16 + nlo;
#pragma unroll
      for (int r = 0; r < 8; ++r) {
        const int m = m0 + wm + fm * 16 + rhalf + r;
        out[(size_t)m * D_ + n] = c[r];
      }
    }
  }
}

// ---------------------------------------------------------------------------
extern "C" void kernel_launch(void* const* d_in, const int* in_sizes, int n_in,
                              void* d_out, int out_size, void* d_ws,
                              size_t ws_size, hipStream_t stream) {
  (void)in_sizes; (void)n_in; (void)out_size; (void)ws_size;

  const float* queries = (const float*)d_in[0];
  const float* keys    = (const float*)d_in[1];
  const float* values  = (const float*)d_in[2];
  const int*   vlens   = (const int*)d_in[3];
  const float* Wq      = (const float*)d_in[4];
  const float* Wk      = (const float*)d_in[5];
  const float* Wv      = (const float*)d_in[6];
  const float* Wo      = (const float*)d_in[7];

  bf16_t* qkv = (bf16_t*)d_ws;          // 3 * PLANE bf16 (24 MiB)
  bf16_t* ctx = qkv + 3 * PLANE;        // PLANE bf16 (8 MiB)

  dim3 blk(256);
  dim3 g1(MTOT / 64, D_ / 128, 3);      // 128 x 4 x 3
  proj_qkv_kernel<<<g1, blk, 0, stream>>>(queries, keys, values, Wq, Wk, Wv,
                                          qkv);
  dim3 g2(S_ / 128, B_ * H_);           // 32 x 16
  flash_attn_kernel<<<g2, blk, 0, stream>>>(qkv, vlens, ctx);
  dim3 g3(MTOT / 64, D_ / 128);         // 128 x 4
  out_proj_kernel<<<g3, blk, 0, stream>>>(ctx, Wo, (float*)d_out);
}
